// ALIGNN_37958920962092
// MI455X (gfx1250) — matrix operand order repaired and checked by reading, not compiled, not run on hardware
//
#include <hip/hip_runtime.h>

// ---------------------------------------------------------------------------
// ALIGNN forward on MI455X (gfx1250): bf16 WMMA GEMMs + fused epilogues.
// ---------------------------------------------------------------------------
#define H        256
#define NNODES   4096
#define NEDGES   49152
#define NTRIP    262144
#define NCONV    12
#define BN_EPS   1e-5f
#define LDST     264          // LDS row stride in ushorts (256 + 8 pad)

typedef __attribute__((ext_vector_type(16))) __bf16 v16bf;
typedef __attribute__((ext_vector_type(8)))  float  v8f;

struct __align__(16) U4 { unsigned int x, y, z, w; };
union AFrag { v16bf v; U4 q[2]; };

__device__ __forceinline__ unsigned short f2bf(float f) {
  unsigned int u = __float_as_uint(f);
  u += 0x7FFFu + ((u >> 16) & 1u);          // round-to-nearest-even
  return (unsigned short)(u >> 16);
}
__device__ __forceinline__ float sigmoidf_(float x) { return 1.0f / (1.0f + __expf(-x)); }
__device__ __forceinline__ float siluf_(float x)    { return x / (1.0f + __expf(-x)); }

// --------------------------- weight prep -----------------------------------
// W[l,w,k,n] (f32) -> WT[l,w,n,k] (bf16): B fragments become contiguous-K loads.
__global__ void k_prep_w(const float* __restrict__ W, unsigned short* __restrict__ WT) {
  long long total = (long long)NCONV * 5 * H * H;
  long long i = (long long)blockIdx.x * blockDim.x + threadIdx.x;
  if (i >= total) return;
  int k = (int)(i % H);
  long long r = i / H;
  int n = (int)(r % H);
  long long mtx = r / H;
  WT[i] = f2bf(W[mtx * H * H + (long long)k * H + n]);
}

// --------------------------- node-side 4-GEMM ------------------------------
// One LDS tile of nf (64x256, bf16) feeds e_src(W0), e_dst(W1), Ah(W3), Bh(W4).
__global__ __launch_bounds__(256) void k_gemm4(
    const float* __restrict__ X,
    const unsigned short* __restrict__ WTl,   // WT + l*5*H*H
    const float* __restrict__ bl,             // b  + l*5*H
    float* __restrict__ O0, float* __restrict__ O1,
    float* __restrict__ O2, float* __restrict__ O3) {
  __shared__ unsigned short Xs[64 * LDST];
  const int tid = threadIdx.x;
  const long long rowBase = (long long)blockIdx.x * 64;

  // stage + convert activation tile (coalesced float4 loads)
  const float4* Xv = (const float4*)(X + rowBase * H);
  #pragma unroll
  for (int it = 0; it < 16; ++it) {
    int idx = tid + it * 256;                 // 0..4095 float4s
    int row = idx >> 6, c4 = idx & 63;
    float4 f = Xv[row * 64 + c4];
    unsigned int lo = (unsigned int)f2bf(f.x) | ((unsigned int)f2bf(f.y) << 16);
    unsigned int hi = (unsigned int)f2bf(f.z) | ((unsigned int)f2bf(f.w) << 16);
    *(uint2*)&Xs[row * LDST + c4 * 4] = make_uint2(lo, hi);
  }
  __syncthreads();

  const int wave = tid >> 5, lane = tid & 31;
  const int wm = wave & 3, wn = wave >> 2;
  const int l16 = lane & 15, hi16 = lane >> 4;
  const int arow = wm * 16 + l16;
  const int koffA = hi16 * 16;                // contiguous-K A-fragment layout (matches B)
  const int koffB = hi16 * 16;                // contiguous-K B-fragment layout (matches A)
  float* outs[4] = {O0, O1, O2, O3};

  for (int w = 0; w < 4; ++w) {
    const int widx = w + (w >> 1);            // {0,1,3,4}
    const unsigned short* Wm = WTl + (long long)widx * H * H;
    const float* bv = bl + widx * H;
    v8f acc[8] = {};
    for (int kk = 0; kk < H; kk += 32) {
      AFrag a;
      a.q[0] = *(const U4*)&Xs[arow * LDST + kk + koffA];
      a.q[1] = *(const U4*)&Xs[arow * LDST + kk + 8 + koffA];
      #pragma unroll
      for (int t = 0; t < 8; ++t) {
        int n = wn * 128 + t * 16 + l16;
        AFrag bf;
        const U4* bp = (const U4*)(Wm + (long long)n * H + kk + koffB);
        bf.q[0] = bp[0]; bf.q[1] = bp[1];
        acc[t] = __builtin_amdgcn_wmma_f32_16x16x32_bf16(
            false, a.v, false, bf.v, (short)0, acc[t], false, false);
      }
    }
    float* O = outs[w];
    const long long rbase = rowBase + wm * 16 + hi16 * 8;
    #pragma unroll
    for (int t = 0; t < 8; ++t) {
      int n = wn * 128 + t * 16 + l16;
      float bb = bv[n];
      #pragma unroll
      for (int r = 0; r < 8; ++r)
        O[(rbase + r) * H + n] = acc[t][r] + bb;
    }
  }
}

// --------------------------- edge GEMM + fused gather/scatter --------------
// m = ef@W2 + b2 + e_src[src] + e_dst[dst];  sigma = sigmoid(m)
// ssh[dst] += sigma * Bh[src];  ss[dst] += sigma;  statE += colsum/colsumsq(m)
__global__ __launch_bounds__(256) void k_gemm_edge(
    const float* __restrict__ EF,
    const unsigned short* __restrict__ W2, const float* __restrict__ B2,
    const float* __restrict__ ESRC, const float* __restrict__ EDST,
    const float* __restrict__ BH,
    const int* __restrict__ src, const int* __restrict__ dst,
    float* __restrict__ Mout, float* __restrict__ SSH, float* __restrict__ SS,
    float* __restrict__ statE) {
  __shared__ unsigned short Xs[64 * LDST];
  __shared__ float colred[512];
  const int tid = threadIdx.x;
  const long long rowBase = (long long)blockIdx.x * 64;

  const float4* Xv = (const float4*)(EF + rowBase * H);
  #pragma unroll
  for (int it = 0; it < 16; ++it) {
    int idx = tid + it * 256;
    int row = idx >> 6, c4 = idx & 63;
    float4 f = Xv[row * 64 + c4];
    unsigned int lo = (unsigned int)f2bf(f.x) | ((unsigned int)f2bf(f.y) << 16);
    unsigned int hi = (unsigned int)f2bf(f.z) | ((unsigned int)f2bf(f.w) << 16);
    *(uint2*)&Xs[row * LDST + c4 * 4] = make_uint2(lo, hi);
  }
  __syncthreads();

  const int wave = tid >> 5, lane = tid & 31;
  const int wm = wave & 3, wn = wave >> 2;
  const int l16 = lane & 15, hi16 = lane >> 4;
  const int arow = wm * 16 + l16;
  const int koffA = hi16 * 16, koffB = hi16 * 16;

  v8f acc[8] = {};
  for (int kk = 0; kk < H; kk += 32) {
    AFrag a;
    a.q[0] = *(const U4*)&Xs[arow * LDST + kk + koffA];
    a.q[1] = *(const U4*)&Xs[arow * LDST + kk + 8 + koffA];
    #pragma unroll
    for (int t = 0; t < 8; ++t) {
      int n = wn * 128 + t * 16 + l16;
      AFrag bf;
      const U4* bp = (const U4*)(W2 + (long long)n * H + kk + koffB);
      bf.q[0] = bp[0]; bf.q[1] = bp[1];
      acc[t] = __builtin_amdgcn_wmma_f32_16x16x32_bf16(
          false, a.v, false, bf.v, (short)0, acc[t], false, false);
    }
  }

  colred[tid] = 0.0f; colred[tid + 256] = 0.0f;
  __syncthreads();

  float s[8], sq[8];
  #pragma unroll
  for (int t = 0; t < 8; ++t) { s[t] = 0.0f; sq[t] = 0.0f; }

  const long long ebase = rowBase + wm * 16 + hi16 * 8;
  for (int r = 0; r < 8; ++r) {
    const long long e = ebase + r;
    const int si = src[e], di = dst[e];
    const float* esr = ESRC + (long long)si * H;
    const float* eds = EDST + (long long)di * H;
    const float* bhr = BH   + (long long)si * H;
    float* sshr = SSH + (long long)di * H;
    float* ssr  = SS  + (long long)di * H;
    float* mr   = Mout + e * H;
    #pragma unroll
    for (int t = 0; t < 8; ++t) {
      int n = wn * 128 + t * 16 + l16;
      float mval = acc[t][r] + B2[n] + esr[n] + eds[n];
      mr[n] = mval;
      float sg = sigmoidf_(mval);
      atomicAdd(sshr + n, sg * bhr[n]);
      atomicAdd(ssr + n, sg);
      s[t] += mval; sq[t] += mval * mval;
    }
  }
  #pragma unroll
  for (int t = 0; t < 8; ++t) {
    int n = wn * 128 + t * 16 + l16;
    atomicAdd(&colred[n], s[t]);            // ds_add_f32
    atomicAdd(&colred[256 + n], sq[t]);
  }
  __syncthreads();
  atomicAdd(&statE[tid], colred[tid]);
  atomicAdd(&statE[256 + tid], colred[256 + tid]);
}

// --------------------------- node update + BN partials ---------------------
__global__ __launch_bounds__(256) void k_node_update(
    const float* __restrict__ AH, const float* __restrict__ SSH,
    const float* __restrict__ SS, float* __restrict__ T,
    float* __restrict__ statN) {
  const int c = threadIdx.x;
  const long long r0 = (long long)blockIdx.x * 64;
  float s = 0.0f, sq = 0.0f;
  for (int r = 0; r < 64; ++r) {
    long long idx = (r0 + r) * H + c;
    float t = AH[idx] + SSH[idx] / (SS[idx] + 1e-6f);
    T[idx] = t;
    s += t; sq += t * t;
  }
  atomicAdd(&statN[c], s);
  atomicAdd(&statN[256 + c], sq);
}

// --------------------------- BN finalize: fold gamma/beta/mu/sigma ---------
__global__ void k_finalize(const float* __restrict__ statN, const float* __restrict__ statE,
                           const float* __restrict__ gamma, const float* __restrict__ beta,
                           float* __restrict__ coefN, float* __restrict__ coefE,
                           int Mn, int Me) {
  const int c = threadIdx.x;
  {
    float mu  = statN[c] / (float)Mn;
    float var = statN[256 + c] / (float)Mn - mu * mu;
    float a   = gamma[c] * rsqrtf(var + BN_EPS);
    coefN[c] = a; coefN[256 + c] = beta[c] - mu * a;
  }
  {
    float mu  = statE[c] / (float)Me;
    float var = statE[256 + c] / (float)Me - mu * mu;
    float a   = gamma[H + c] * rsqrtf(var + BN_EPS);
    coefE[c] = a; coefE[256 + c] = beta[H + c] - mu * a;
  }
}

// --------------------------- residual: F += silu(V*a + c) ------------------
__global__ __launch_bounds__(256) void k_residual(
    float* __restrict__ F, const float* __restrict__ V, const float* __restrict__ coef) {
  const int c = threadIdx.x;
  const float a = coef[c], cc = coef[256 + c];
  const long long r0 = (long long)blockIdx.x * 64;
  for (int r = 0; r < 64; ++r) {
    long long idx = (r0 + r) * H + c;
    F[idx] += siluf_(V[idx] * a + cc);
  }
}

// --------------------------- host orchestration ----------------------------
static void run_egconv(float* nf, int nM, float* ef, int eM,
                       const int* src, const int* dst, int l,
                       const float* b, const float* bng, const float* bnb,
                       unsigned short* WT, float* esrc, float* edst,
                       float* ah, float* bh, float* mbuf, float* ssh, float* ss,
                       float* statN, float* statE, float* coefN, float* coefE,
                       hipStream_t stream) {
  hipMemsetAsync(ssh, 0, (size_t)nM * H * sizeof(float), stream);
  hipMemsetAsync(ss,  0, (size_t)nM * H * sizeof(float), stream);
  hipMemsetAsync(statN, 0, 512 * sizeof(float), stream);
  hipMemsetAsync(statE, 0, 512 * sizeof(float), stream);

  k_gemm4<<<nM / 64, 256, 0, stream>>>(
      nf, WT + (size_t)l * 5 * H * H, b + (size_t)l * 5 * H, esrc, edst, ah, bh);
  k_gemm_edge<<<eM / 64, 256, 0, stream>>>(
      ef, WT + ((size_t)l * 5 + 2) * H * H, b + ((size_t)l * 5 + 2) * H,
      esrc, edst, bh, src, dst, mbuf, ssh, ss, statE);
  // t-buffer aliases esrc (dead after k_gemm_edge)
  k_node_update<<<nM / 64, 256, 0, stream>>>(ah, ssh, ss, esrc, statN);
  k_finalize<<<1, 256, 0, stream>>>(statN, statE,
      bng + (size_t)l * 2 * H, bnb + (size_t)l * 2 * H, coefN, coefE, nM, eM);
  k_residual<<<nM / 64, 256, 0, stream>>>(nf, esrc, coefN);
  k_residual<<<eM / 64, 256, 0, stream>>>(ef, mbuf, coefE);
}

extern "C" void kernel_launch(void* const* d_in, const int* in_sizes, int n_in,
                              void* d_out, int out_size, void* d_ws, size_t ws_size,
                              hipStream_t stream) {
  const float* x   = (const float*)d_in[0];
  const float* y   = (const float*)d_in[1];
  const float* z   = (const float*)d_in[2];
  const float* W   = (const float*)d_in[3];
  const float* b   = (const float*)d_in[4];
  const float* bng = (const float*)d_in[5];
  const float* bnb = (const float*)d_in[6];
  const int* src  = (const int*)d_in[7];
  const int* dst  = (const int*)d_in[8];
  const int* lsrc = (const int*)d_in[9];
  const int* ldst = (const int*)d_in[10];

  char* p = (char*)d_ws;
  auto alloc = [&](size_t bytes) -> char* {
    char* r = p; p += (bytes + 255) & ~(size_t)255; return r;
  };
  unsigned short* WT = (unsigned short*)alloc((size_t)NCONV * 5 * H * H * 2);
  float* xb   = (float*)alloc((size_t)NNODES * H * 4);
  float* yb   = (float*)alloc((size_t)NEDGES * H * 4);
  float* zb   = (float*)alloc((size_t)NTRIP  * H * 4);
  float* esrc = (float*)alloc((size_t)NEDGES * H * 4);   // also t-buffer
  float* edst = (float*)alloc((size_t)NEDGES * H * 4);
  float* ah   = (float*)alloc((size_t)NEDGES * H * 4);
  float* bh   = (float*)alloc((size_t)NEDGES * H * 4);
  float* mbuf = (float*)alloc((size_t)NTRIP  * H * 4);
  float* ssh  = (float*)alloc((size_t)NEDGES * H * 4);
  float* ss   = (float*)alloc((size_t)NEDGES * H * 4);
  float* statN = (float*)alloc(512 * 4);
  float* statE = (float*)alloc(512 * 4);
  float* coefN = (float*)alloc(512 * 4);
  float* coefE = (float*)alloc(512 * 4);

  long long wtot = (long long)NCONV * 5 * H * H;
  k_prep_w<<<(int)((wtot + 255) / 256), 256, 0, stream>>>(W, WT);

  hipMemcpyAsync(xb, x, (size_t)NNODES * H * 4, hipMemcpyDeviceToDevice, stream);
  hipMemcpyAsync(yb, y, (size_t)NEDGES * H * 4, hipMemcpyDeviceToDevice, stream);
  hipMemcpyAsync(zb, z, (size_t)NTRIP  * H * 4, hipMemcpyDeviceToDevice, stream);

  // 4 ALIGNN layers: crystal-graph conv then line-graph conv
  for (int i = 0; i < 4; ++i) {
    run_egconv(xb, NNODES, yb, NEDGES, src, dst, 2 * i, b, bng, bnb,
               WT, esrc, edst, ah, bh, mbuf, ssh, ss, statN, statE, coefN, coefE, stream);
    run_egconv(yb, NEDGES, zb, NTRIP, lsrc, ldst, 2 * i + 1, b, bng, bnb,
               WT, esrc, edst, ah, bh, mbuf, ssh, ss, statN, statE, coefN, coefE, stream);
  }
  // 4 gcn layers on crystal graph
  for (int l = 8; l < 12; ++l) {
    run_egconv(xb, NNODES, yb, NEDGES, src, dst, l, b, bng, bnb,
               WT, esrc, edst, ah, bh, mbuf, ssh, ss, statN, statE, coefN, coefE, stream);
  }

  hipMemcpyAsync(d_out, xb, (size_t)NNODES * H * 4, hipMemcpyDeviceToDevice, stream);
}
